// EncoderBlock_60868276519509
// MI455X (gfx1250) — matrix operand order repaired and checked by reading, not compiled, not run on hardware
//
#include <hip/hip_runtime.h>
#include <math.h>

// ---------------------------------------------------------------------------
// MI455X / gfx1250 transformer encoder block.
//  - all matmuls: v_wmma_f32_16x16x32_bf16 (wave32, fp32 accumulate)
//  - GEMM tiles: Tensor Data Mover (tensor_load_to_lds) double-buffered via
//    TENSORcnt; LDS padding done by the TDM pad feature (stride 40 bf16)
//  - all LDS->VGPR fragment gathers are 2x ds_load_b128 per lane
// ---------------------------------------------------------------------------

typedef __attribute__((ext_vector_type(16))) __bf16        bf16x16;
typedef __attribute__((ext_vector_type(8)))  float         f32x8;
typedef __attribute__((ext_vector_type(4)))  unsigned int  u32x4;
typedef __attribute__((ext_vector_type(8)))  int           i32x8;
typedef __attribute__((ext_vector_type(4)))  int           i32x4;

#define EMB   768
#define HEADS 8
#define HD    96
#define BB    8
#define NN    1024
#define ROWS  (BB * NN)   // 8192

__device__ __forceinline__ f32x8 wmma_bf16(bf16x16 a, bf16x16 b, f32x8 c) {
  return __builtin_amdgcn_wmma_f32_16x16x32_bf16(
      false, a, false, b, (short)0, c, false, false);
}

// A fragment (16x32, row-major [M][K] tile, stride multiple of 8 elements):
// per lane two contiguous 16B runs at k0+half*8 and k0+16+half*8.
__device__ __forceinline__ bf16x16 frag_a(const __bf16* base, int stride,
                                          int m0, int k0, int lane) {
  const int m = m0 + (lane & 15), half = lane >> 4;
  const __bf16* p = base + m * stride + k0 + half * 8;
  union { u32x4 q[2]; bf16x16 v; } t;
  t.q[0] = *(const u32x4*)p;
  t.q[1] = *(const u32x4*)(p + 16);
  return t.v;
}

// B fragment from an [N][K] (transposed) tile: per lane two contiguous 16B runs
// at row n = n0+lane%16, cols k0+half*8 and k0+16+half*8.
__device__ __forceinline__ bf16x16 frag_bt(const __bf16* base, int stride,
                                           int k0, int n0, int lane) {
  const int n = n0 + (lane & 15), half = lane >> 4;
  const __bf16* p = base + n * stride + k0 + half * 8;
  union { u32x4 q[2]; bf16x16 v; } t;
  t.q[0] = *(const u32x4*)p;
  t.q[1] = *(const u32x4*)(p + 16);
  return t.v;
}

// ---------------------------------------------------------------------------
// Tensor Data Mover: 2D tile (tile_d0 x tile_d1, 2-byte elems) -> LDS,
// LDS rows padded 32 -> 40 bf16 (pad_interval=16 DW (code 3), amount=4 DW (code 3)).
// D# per CDNA5 ISA ch.8: group0 {count=1, lds_addr, global_addr, type=2},
// group1 {data_size=2B, pad cfg, dims, tile dims, stride}.
// This toolchain exposes the 6-arg builtin form (extra int32x8 group, then cpol).
// ---------------------------------------------------------------------------
__device__ __forceinline__ void tdm_load_tile(unsigned lds_off, const __bf16* gptr,
                                              unsigned tensor_d0, unsigned tensor_d1,
                                              unsigned tile_d0, unsigned tile_d1,
                                              unsigned stride0) {
  unsigned long long ga = (unsigned long long)(size_t)gptr;
  u32x4 g0;
  g0[0] = 1u;                                   // count=1 (user descriptor)
  g0[1] = lds_off;                              // LDS byte address
  g0[2] = (unsigned)ga;                         // global_addr[31:0]
  g0[3] = (unsigned)((ga >> 32) & 0x01FFFFFFu) | 0x80000000u;  // addr[56:32] | type=2
  i32x8 g1;
  g1[0] = (int)((1u << 16) | (1u << 20) | (3u << 22) | (3u << 25)); // 2B, pad on
  g1[1] = (int)((tensor_d0 & 0xFFFFu) << 16);
  g1[2] = (int)((tensor_d0 >> 16) | ((tensor_d1 & 0xFFFFu) << 16));
  g1[3] = (int)((tensor_d1 >> 16) | (tile_d0 << 16));
  g1[4] = (int)(tile_d1 & 0xFFFFu);
  g1[5] = (int)stride0;                         // tensor_dim0_stride[31:0]
  g1[6] = 0;
  g1[7] = 0;
  i32x4 z4; z4[0] = z4[1] = z4[2] = z4[3] = 0;
  i32x8 z8;
#pragma unroll
  for (int i = 0; i < 8; ++i) z8[i] = 0;
  __builtin_amdgcn_tensor_load_to_lds(g0, g1, z4, z4, z8, 0);
}

// ---------------------------------------------------------------------------
// fp32 -> bf16 elementwise (x, activations)
// ---------------------------------------------------------------------------
__global__ void __launch_bounds__(256)
cvt_bf16_kernel(const float* __restrict__ in, __bf16* __restrict__ out) {
  int base = blockIdx.x * 2048 + threadIdx.x;
#pragma unroll
  for (int j = 0; j < 8; ++j) out[base + j * 256] = (__bf16)in[base + j * 256];
}

// fp32 [R][C] -> bf16 [C][R] (weight transpose+convert, once per launch)
__global__ void __launch_bounds__(256)
transpose_cvt_kernel(const float* __restrict__ in, __bf16* __restrict__ out,
                     int R, int C) {
  __shared__ float t[32][33];
  const int bx = blockIdx.x * 32, by = blockIdx.y * 32;
  const int tx = threadIdx.x & 31, ty = threadIdx.x >> 5;
#pragma unroll
  for (int i = 0; i < 32; i += 8)
    t[ty + i][tx] = in[(size_t)(by + ty + i) * C + bx + tx];
  __syncthreads();
#pragma unroll
  for (int i = 0; i < 32; i += 8)
    out[(size_t)(bx + ty + i) * R + by + tx] = (__bf16)t[tx][ty + i];
}

// ---------------------------------------------------------------------------
// GEMM: out[M,N] = epi(A[M,K] @ BT[N,K]^T + bias).  A, BT are bf16 global.
// Block tile 128x128, 8 waves (wave tile 32x64), K-step 32.
// Both tiles DMA'd by TDM, double-buffered on TENSORcnt.
// EPI: 0=bias, 1=bias+exact GELU, 2=bias+residual.
// ---------------------------------------------------------------------------
#define TS 40   // padded LDS tile stride (bf16 elements), produced by TDM pads

template <typename OT, int EPI>
__global__ void __launch_bounds__(256)
gemm_tdm(const __bf16* __restrict__ A, const __bf16* __restrict__ BT,
         const float* __restrict__ bias, const float* __restrict__ resid,
         OT* __restrict__ out, int M, int N, int K) {
  __shared__ __bf16 As[2][128 * TS];
  __shared__ __bf16 Bs[2][128 * TS];

  const int tid  = threadIdx.x;
  const int lane = tid & 31;
  const int w    = tid >> 5;
  const int wm   = w & 3;
  const int wn   = w >> 2;
  const int bm   = blockIdx.y * 128;
  const int bn   = blockIdx.x * 128;
  const int nk   = K >> 5;

  f32x8 acc[2][4];
#pragma unroll
  for (int i = 0; i < 2; ++i)
#pragma unroll
    for (int j = 0; j < 4; ++j)
#pragma unroll
      for (int e = 0; e < 8; ++e) acc[i][j][e] = 0.0f;

  if (w == 0) {
    tdm_load_tile((unsigned)(size_t)&As[0][0], A + (size_t)bm * K,
                  (unsigned)K, (unsigned)M, 32u, 128u, (unsigned)K);
    tdm_load_tile((unsigned)(size_t)&Bs[0][0], BT + (size_t)bn * K,
                  (unsigned)K, (unsigned)N, 32u, 128u, (unsigned)K);
  }

  for (int it = 0; it < nk; ++it) {
    const int cur = it & 1;
    if (w == 0) __builtin_amdgcn_s_wait_tensorcnt(0);
    __syncthreads();   // cur tiles ready; prefetch buffer free (computed last iter)
    if (w == 0 && it + 1 < nk) {
      const int kt = (it + 1) << 5;
      tdm_load_tile((unsigned)(size_t)&As[cur ^ 1][0], A + (size_t)bm * K + kt,
                    (unsigned)K, (unsigned)M, 32u, 128u, (unsigned)K);
      tdm_load_tile((unsigned)(size_t)&Bs[cur ^ 1][0], BT + (size_t)bn * K + kt,
                    (unsigned)K, (unsigned)N, 32u, 128u, (unsigned)K);
    }
    bf16x16 af[2], bfv[4];
#pragma unroll
    for (int tm = 0; tm < 2; ++tm)
      af[tm] = frag_a(As[cur], TS, wm * 32 + tm * 16, 0, lane);
#pragma unroll
    for (int tn = 0; tn < 4; ++tn)
      bfv[tn] = frag_bt(Bs[cur], TS, 0, wn * 64 + tn * 16, lane);
#pragma unroll
    for (int tm = 0; tm < 2; ++tm)
#pragma unroll
      for (int tn = 0; tn < 4; ++tn)
        acc[tm][tn] = wmma_bf16(af[tm], bfv[tn], acc[tm][tn]);
  }

  const int half = lane >> 4;
#pragma unroll
  for (int tm = 0; tm < 2; ++tm)
#pragma unroll
    for (int tn = 0; tn < 4; ++tn)
#pragma unroll
      for (int r = 0; r < 8; ++r) {
        int row = bm + wm * 32 + tm * 16 + half * 8 + r;
        int col = bn + wn * 64 + tn * 16 + (lane & 15);
        float v = acc[tm][tn][r] + bias[col];
        if (EPI == 1) v = 0.5f * v * (1.0f + erff(v * 0.70710678118654752f));
        if (EPI == 2) v += resid[(size_t)row * N + col];
        out[(size_t)row * N + col] = (OT)v;
      }
}

// ---------------------------------------------------------------------------
// Flash attention, one block per (b, h, 128 query rows).
// qkvr bf16 [B*N, 3072], column = h*384 + d*4 + {q,k,v,r}.
// Quirk: softmax of UNSCALED logits, * 1/sqrt(EMB) after, then * r-gate.
// ---------------------------------------------------------------------------
#define QKS 104   // Q/K LDS stride (rows x 96 cols used)
#define VTS 136   // Vt / P LDS stride

__global__ void __launch_bounds__(256)
attn_kernel(const __bf16* __restrict__ qkvr, __bf16* __restrict__ out) {
  extern __shared__ __bf16 smem[];
  __bf16* Qs = smem;                   // 128 x QKS   (rows = q, cols = d)
  __bf16* Ks = Qs + 128 * QKS;         // 128 x QKS   (rows = kt, cols = d)
  __bf16* Vt = Ks + 128 * QKS;         // 96 x VTS    (rows = d, cols = kt)
  __bf16* Ps = Vt + 96 * VTS;          // 8 waves x 16 x VTS

  const int tid  = threadIdx.x;
  const int lane = tid & 31;
  const int w    = tid >> 5;
  const int half = lane >> 4;
  const int qb   = blockIdx.x & 7;
  const int bh   = blockIdx.x >> 3;
  const int b    = bh >> 3;
  const int h    = bh & 7;

  for (int i = 0; i < 48; ++i) {
    int idx = i * 256 + tid;
    int r = idx / 96, c = idx - r * 96;
    Qs[r * QKS + c] =
        qkvr[(size_t)(b * NN + qb * 128 + r) * 3072 + h * 384 + c * 4 + 0];
  }

  float rmax[8], rsum[8];
#pragma unroll
  for (int r = 0; r < 8; ++r) { rmax[r] = -INFINITY; rsum[r] = 0.0f; }
  f32x8 oacc[6];
#pragma unroll
  for (int ot = 0; ot < 6; ++ot)
#pragma unroll
    for (int e = 0; e < 8; ++e) oacc[ot][e] = 0.0f;

  for (int kb = 0; kb < 8; ++kb) {
    __syncthreads();
    for (int i = 0; i < 48; ++i) {
      int idx = i * 256 + tid;
      int r = idx / 96, c = idx - r * 96;
      size_t g = (size_t)(b * NN + kb * 128 + r) * 3072 + h * 384 + (size_t)c * 4;
      Ks[r * QKS + c] = qkvr[g + 1];
      Vt[c * VTS + r] = qkvr[g + 2];   // transposed: rows=d, cols=kt
    }
    __syncthreads();

    // S (16 q-rows x 128 kt) = Q @ K^T;  Ks is [kt][d] -> frag_bt gives K^T
    f32x8 sacc[8];
#pragma unroll
    for (int nt = 0; nt < 8; ++nt)
#pragma unroll
      for (int e = 0; e < 8; ++e) sacc[nt][e] = 0.0f;
#pragma unroll
    for (int ks = 0; ks < 3; ++ks) {
      bf16x16 aq = frag_a(Qs, QKS, w * 16, ks * 32, lane);
#pragma unroll
      for (int nt = 0; nt < 8; ++nt) {
        bf16x16 bk = frag_bt(Ks, QKS, ks * 32, nt * 16, lane);
        sacc[nt] = wmma_bf16(aq, bk, sacc[nt]);
      }
    }

    // Online softmax (rows live per 16-lane half; xor-reduce 1,2,4,8)
#pragma unroll
    for (int r = 0; r < 8; ++r) {
      float tmax = sacc[0][r];
#pragma unroll
      for (int nt = 1; nt < 8; ++nt) tmax = fmaxf(tmax, sacc[nt][r]);
      tmax = fmaxf(tmax, __shfl_xor(tmax, 1, 32));
      tmax = fmaxf(tmax, __shfl_xor(tmax, 2, 32));
      tmax = fmaxf(tmax, __shfl_xor(tmax, 4, 32));
      tmax = fmaxf(tmax, __shfl_xor(tmax, 8, 32));
      float mnew = fmaxf(rmax[r], tmax);
      float corr = __expf(rmax[r] - mnew);
      rmax[r] = mnew;
      float ps = 0.0f;
#pragma unroll
      for (int nt = 0; nt < 8; ++nt) {
        float p = __expf(sacc[nt][r] - mnew);
        sacc[nt][r] = p;
        ps += p;
      }
      ps += __shfl_xor(ps, 1, 32);
      ps += __shfl_xor(ps, 2, 32);
      ps += __shfl_xor(ps, 4, 32);
      ps += __shfl_xor(ps, 8, 32);
      rsum[r] = rsum[r] * corr + ps;
#pragma unroll
      for (int ot = 0; ot < 6; ++ot) oacc[ot][r] *= corr;
    }

    // Stage P (16 x 128) per-wave to re-layout C -> A fragments
    __bf16* Pw = Ps + w * 16 * VTS;
#pragma unroll
    for (int nt = 0; nt < 8; ++nt)
#pragma unroll
      for (int r = 0; r < 8; ++r)
        Pw[(half * 8 + r) * VTS + nt * 16 + (lane & 15)] = (__bf16)sacc[nt][r];
    __syncthreads();

    // O += P @ V   (Vt is [d][kt] -> frag_bt gives V in B layout)
#pragma unroll
    for (int kk = 0; kk < 4; ++kk) {
      bf16x16 ap = frag_a(Pw, VTS, 0, kk * 32, lane);
#pragma unroll
      for (int ot = 0; ot < 6; ++ot) {
        bf16x16 bv = frag_bt(Vt, VTS, kk * 32, ot * 16, lane);
        oacc[ot] = wmma_bf16(ap, bv, oacc[ot]);
      }
    }
  }

  const float inv_sqrt_e = 0.03608439182435161f;  // 1/sqrt(768)
#pragma unroll
  for (int ot = 0; ot < 6; ++ot)
#pragma unroll
    for (int r = 0; r < 8; ++r) {
      int qrow = qb * 128 + w * 16 + half * 8 + r;
      int dcol = ot * 16 + (lane & 15);
      float rg = (float)qkvr[(size_t)(b * NN + qrow) * 3072 + h * 384 + dcol * 4 + 3];
      float v = oacc[ot][r] / rsum[r] * inv_sqrt_e * rg;
      out[(size_t)(b * NN + qrow) * EMB + h * HD + dcol] = (__bf16)v;
    }
}

// ---------------------------------------------------------------------------
// LayerNorm (width 768), optional bf16 mirror for the next GEMM's TDM A-feed
// ---------------------------------------------------------------------------
__global__ void __launch_bounds__(256)
ln_kernel(const float* __restrict__ x, const float* __restrict__ g,
          const float* __restrict__ beta, float* __restrict__ out,
          __bf16* __restrict__ obf) {
  __shared__ float s1[256], s2[256];
  const int row = blockIdx.x, tid = threadIdx.x;
  const float* xr = x + (size_t)row * EMB;
  float a0 = xr[tid], a1 = xr[tid + 256], a2 = xr[tid + 512];
  s1[tid] = a0 + a1 + a2;
  s2[tid] = a0 * a0 + a1 * a1 + a2 * a2;
  __syncthreads();
  for (int s = 128; s > 0; s >>= 1) {
    if (tid < s) { s1[tid] += s1[tid + s]; s2[tid] += s2[tid + s]; }
    __syncthreads();
  }
  float mu  = s1[0] * (1.0f / EMB);
  float var = s2[0] * (1.0f / EMB) - mu * mu;
  float rs  = rsqrtf(var + 1e-5f);
  float v0 = (a0 - mu) * rs * g[tid]       + beta[tid];
  float v1 = (a1 - mu) * rs * g[tid + 256] + beta[tid + 256];
  float v2 = (a2 - mu) * rs * g[tid + 512] + beta[tid + 512];
  float* o = out + (size_t)row * EMB;
  o[tid] = v0; o[tid + 256] = v1; o[tid + 512] = v2;
  if (obf) {
    __bf16* ob = obf + (size_t)row * EMB;
    ob[tid] = (__bf16)v0; ob[tid + 256] = (__bf16)v1; ob[tid + 512] = (__bf16)v2;
  }
}

// ---------------------------------------------------------------------------
extern "C" void kernel_launch(void* const* d_in, const int* in_sizes, int n_in,
                              void* d_out, int out_size, void* d_ws, size_t ws_size,
                              hipStream_t stream) {
  (void)in_sizes; (void)n_in; (void)out_size; (void)ws_size;
  const float* x      = (const float*)d_in[0];
  const float* w_qkvr = (const float*)d_in[1];
  const float* b_qkvr = (const float*)d_in[2];
  const float* w_proj = (const float*)d_in[3];
  const float* b_proj = (const float*)d_in[4];
  const float* ln1_g  = (const float*)d_in[5];
  const float* ln1_b  = (const float*)d_in[6];
  const float* w_ff1  = (const float*)d_in[7];
  const float* b_ff1  = (const float*)d_in[8];
  const float* w_ff2  = (const float*)d_in[9];
  const float* b_ff2  = (const float*)d_in[10];
  const float* ln2_g  = (const float*)d_in[11];
  const float* ln2_b  = (const float*)d_in[12];

  char* ws = (char*)d_ws;
  __bf16* qkvr   = (__bf16*)(ws + 0);                  // 48 MB (reused as hbuf)
  __bf16* hbuf   = (__bf16*)(ws + 0);
  __bf16* attnb  = (__bf16*)(ws + (size_t)50331648);   // 12 MB
  float*  pre    = (float*) (ws + (size_t)62914560);   // 24 MB
  float*  x1     = (float*) (ws + (size_t)88080384);   // 24 MB
  __bf16* x1b    = (__bf16*)(ws + (size_t)113246208);  // 12 MB
  __bf16* xb     = (__bf16*)(ws + (size_t)125829120);  // 12 MB
  __bf16* wqkvrT = (__bf16*)(ws + (size_t)138412032);  // [3072][768]
  __bf16* wprojT = (__bf16*)(ws + (size_t)143130624);  // [768][768]
  __bf16* wff1T  = (__bf16*)(ws + (size_t)144310272);  // [3072][768]
  __bf16* wff2T  = (__bf16*)(ws + (size_t)149028864);  // [768][3072]
  float*  outp   = (float*)d_out;

  dim3 blk(256);

  // --- one-time converts: bf16 activations + transposed bf16 weights [N][K]
  cvt_bf16_kernel<<<dim3(ROWS * EMB / 2048), blk, 0, stream>>>(x, xb);
  transpose_cvt_kernel<<<dim3(96, 24), blk, 0, stream>>>(w_qkvr, wqkvrT, EMB, 3072);
  transpose_cvt_kernel<<<dim3(24, 24), blk, 0, stream>>>(w_proj, wprojT, EMB, EMB);
  transpose_cvt_kernel<<<dim3(96, 24), blk, 0, stream>>>(w_ff1, wff1T, EMB, 3072);
  transpose_cvt_kernel<<<dim3(24, 96), blk, 0, stream>>>(w_ff2, wff2T, 3072, EMB);

  // 1) qkvr = x @ w_qkvr + b        (M=8192, N=3072, K=768)
  gemm_tdm<__bf16, 0><<<dim3(24, 64), blk, 0, stream>>>(
      xb, wqkvrT, b_qkvr, nullptr, qkvr, ROWS, 3072, EMB);

  // 2) flash attention + r-gate -> bf16 [B,N,EMB]
  size_t smem = (size_t)(2 * 128 * QKS + 96 * VTS + 8 * 16 * VTS) * sizeof(__bf16);
  attn_kernel<<<dim3(512), blk, smem, stream>>>(qkvr, attnb);

  // 3) pre = attn @ w_proj + b + x  (residual)
  gemm_tdm<float, 2><<<dim3(6, 64), blk, 0, stream>>>(
      attnb, wprojT, b_proj, x, pre, ROWS, EMB, EMB);

  // 4) x1 = LN1(pre)  (+ bf16 mirror)
  ln_kernel<<<dim3(ROWS), blk, 0, stream>>>(pre, ln1_g, ln1_b, x1, x1b);

  // 5) hbuf = gelu(x1 @ w_ff1 + b)
  gemm_tdm<__bf16, 1><<<dim3(24, 64), blk, 0, stream>>>(
      x1b, wff1T, b_ff1, nullptr, hbuf, ROWS, 3072, EMB);

  // 6) pre = hbuf @ w_ff2 + b + x1  (residual)
  gemm_tdm<float, 2><<<dim3(6, 64), blk, 0, stream>>>(
      hbuf, wff2T, b_ff2, x1, pre, ROWS, EMB, 3072);

  // 7) out = LN2(pre)
  ln_kernel<<<dim3(ROWS), blk, 0, stream>>>(pre, ln2_g, ln2_b, outp, nullptr);
}
